// CUDAAttentionLayer_32169305047706
// MI455X (gfx1250) — matrix operand order repaired and checked by reading, not compiled, not run on hardware
//
#include <hip/hip_runtime.h>

// ---------------------------------------------------------------------------
// CDNA5 (gfx1250, wave32) attention pipeline, all-bf16 WMMA GEMMs fed by the
// Tensor Data Mover:
//   - one-shot fp32->bf16 conversion of x and weights
//   - GEMM: TDM (tensor_load_to_lds) DMAs 128x32 bf16 tiles into LDS with
//     80B padded row pitch (bank-conflict-free), double-buffered against
//     TENSORcnt; waves read fragments via ds_load_b128 and issue
//     v_wmma_f32_16x16x32_bf16 (64x64 wave tile, 16 WMMA / K-step)
//   - fp32 row softmax, bf16 in place
// ---------------------------------------------------------------------------

typedef __attribute__((ext_vector_type(16))) __bf16 v16bf;
typedef __attribute__((ext_vector_type(8)))  __bf16 v8bf;
typedef __attribute__((ext_vector_type(8)))  float  v8f;
typedef __attribute__((ext_vector_type(4)))  unsigned int v4u;
typedef __attribute__((ext_vector_type(8)))  int v8i;
typedef __attribute__((ext_vector_type(4)))  int v4i;

#define TILE_ROWS 128
#define TILE_K    32
#define ROW_BYTES 80                          // 64B data + 16B TDM pad
#define TILE_BYTES (TILE_ROWS * ROW_BYTES)    // 10240 B per tile

// ---------------------------------------------------------------------------
// TDM 2-D tile load: 128 rows x 32 bf16 from a row-major [rows, stride]
// matrix into LDS at 80B row pitch.  D# per CDNA5 ISA ch.8 (groups 0/1;
// groups 2/3 zero => 2-D tensor).  Issued once per wave; EXEC ignored.
// ---------------------------------------------------------------------------
__device__ __forceinline__ void tdm_load_tile(unsigned lds_addr,
                                              const __bf16* gsrc,
                                              int row_stride_elems) {
  const unsigned long long ga = (unsigned long long)gsrc;
  v4u g0;
  g0[0] = 1u;                                    // count=1 (valid descriptor)
  g0[1] = lds_addr;                              // LDS byte address
  g0[2] = (unsigned)ga;                          // global_addr[31:0]
  g0[3] = (unsigned)((ga >> 32) & 0x01FFFFFFu)   // global_addr[56:32]
        | (2u << 30);                            // type = 2
  const unsigned s = (unsigned)row_stride_elems;
  v8i g1;
  g1[0] = (int)((1u << 16)     // data_size = 2 bytes
              | (1u << 20)     // pad_enable
              | (3u << 22)     // pad_interval = 16 DWORDs (one 64B row)
              | (3u << 25));   // pad_amount   = 4 DWORDs (16B) -> 80B pitch
  g1[1] = (int)((s & 0xFFFFu) << 16);                      // tensor_dim0 lo16
  g1[2] = (int)(((s >> 16) & 0xFFFFu) | (0xFFFFu << 16));  // dim0 hi | dim1 lo
  g1[3] = (int)((unsigned)TILE_K << 16);                   // dim1 hi | tile_dim0
  g1[4] = (int)TILE_ROWS;                                  // tile_dim1 | tile_dim2=0
  g1[5] = (int)s;                                          // dim0_stride lo32
  g1[6] = 0;                                               // dim0_stride hi16
  g1[7] = 0;
  const v4i zero4 = {0, 0, 0, 0};
  const v8i zero8 = {0, 0, 0, 0, 0, 0, 0, 0};
  __builtin_amdgcn_tensor_load_to_lds(g0, g1, zero4, zero4, zero8, 0);
}

// ---------------------------------------------------------------------------
// Elementwise fp32 -> bf16 (8 elements/thread, b128 in / b128 out).
// ---------------------------------------------------------------------------
__global__ __launch_bounds__(256) void cvt_f32_bf16(
    const float* __restrict__ src, __bf16* __restrict__ dst) {
  const size_t i = ((size_t)blockIdx.x * 256 + threadIdx.x) * 8;
  const float4 a = *(const float4*)(src + i);
  const float4 b = *(const float4*)(src + i + 4);
  v8bf o;
  o[0] = (__bf16)a.x; o[1] = (__bf16)a.y; o[2] = (__bf16)a.z; o[3] = (__bf16)a.w;
  o[4] = (__bf16)b.x; o[5] = (__bf16)b.y; o[6] = (__bf16)b.z; o[7] = (__bf16)b.w;
  *(v8bf*)(dst + i) = o;
}

// ---------------------------------------------------------------------------
// C[m,n] = sum_k A[m,k] * B[n,k]  (+ bias[n]); bf16 operands, fp32 accum.
// Block = 4 waves (2x2) -> 128x128 tile; wave tile 64x64 (4x4 fragments).
// TDM double-buffers 128x32 A/B slabs in LDS; fragments via ds_load_b128.
// ---------------------------------------------------------------------------
template <bool OUT_BF16, bool TRANS_OUT, bool HAS_BIAS>
__global__ __launch_bounds__(128) void wmma_gemm_nt(
    const __bf16* __restrict__ A, const __bf16* __restrict__ B,
    const float* __restrict__ bias, void* __restrict__ Cvoid,
    int K, int lda, int ldb, int ldc,
    long long strideA, long long strideB, long long strideC) {
  __shared__ __align__(16) char smem[4 * TILE_BYTES];  // [buf][A|B], 40 KB
  const int lane = threadIdx.x & 31;
  const int wave = threadIdx.x >> 5;
  const int wm = wave >> 1, wn = wave & 1;
  const long long bz = blockIdx.z;

  const __bf16* Ag = A + bz * strideA + (size_t)(blockIdx.x * 128) * lda;
  const __bf16* Bg = B + bz * strideB + (size_t)(blockIdx.y * 128) * ldb;

  // prologue: DMA first slabs into buffer 0
  if (wave == 0) {
    const unsigned base = (unsigned)(unsigned long long)(void*)&smem[0];
    tdm_load_tile(base, Ag, lda);
    tdm_load_tile(base + TILE_BYTES, Bg, ldb);
    __builtin_amdgcn_s_wait_tensorcnt(0);
  }
  __syncthreads();

  // per-lane LDS fragment offsets (wave32 WMMA layouts, ISA 7.12.2)
  const int arow = wm * 64 + (lane & 15);        // + i*16
  const int ac0  = ((lane >> 4) << 3) * 2;       // bytes: 0 | 16 (K 0..7 / 8..15)
  const int brow = wn * 64 + (lane & 15);        // + j*16
  const int bc0  = ((lane >> 4) << 3) * 2;       // bytes: 0 | 16 (K 0..7 / 8..15)

  v8f acc[4][4];
#pragma unroll
  for (int i = 0; i < 4; ++i)
#pragma unroll
    for (int j = 0; j < 4; ++j)
#pragma unroll
      for (int r = 0; r < 8; ++r) acc[i][j][r] = 0.0f;

  for (int k0 = 0; k0 < K; k0 += 32) {
    const int buf = (k0 >> 5) & 1;
    if (wave == 0 && k0 + 32 < K) {  // DMA next slabs while we compute
      const unsigned base =
          (unsigned)(unsigned long long)(void*)&smem[(buf ^ 1) * (2 * TILE_BYTES)];
      tdm_load_tile(base, Ag + (k0 + 32), lda);
      tdm_load_tile(base + TILE_BYTES, Bg + (k0 + 32), ldb);
    }
    const char* at = &smem[buf * (2 * TILE_BYTES)];
    const char* bt = at + TILE_BYTES;

    v16bf afr[4];
#pragma unroll
    for (int i = 0; i < 4; ++i) {
      const char* p = at + (size_t)(arow + i * 16) * ROW_BYTES + ac0;
      v8bf lo = *(const v8bf*)p;            // K [k..k+7]
      v8bf hi = *(const v8bf*)(p + 32);     // K [k+16..k+23]
#pragma unroll
      for (int t = 0; t < 8; ++t) { afr[i][t] = lo[t]; afr[i][t + 8] = hi[t]; }
    }
    v16bf bfr[4];
#pragma unroll
    for (int j = 0; j < 4; ++j) {
      const char* p = bt + (size_t)(brow + j * 16) * ROW_BYTES + bc0;
      v8bf lo = *(const v8bf*)p;            // K [k..k+7]
      v8bf hi = *(const v8bf*)(p + 32);     // K [k+16..k+23]
#pragma unroll
      for (int t = 0; t < 8; ++t) { bfr[j][t] = lo[t]; bfr[j][t + 8] = hi[t]; }
    }
#pragma unroll
    for (int i = 0; i < 4; ++i)
#pragma unroll
      for (int j = 0; j < 4; ++j)
        acc[i][j] = __builtin_amdgcn_wmma_f32_16x16x32_bf16(
            false, afr[i], false, bfr[j], (short)0, acc[i][j], false, false);

    if (wave == 0) __builtin_amdgcn_s_wait_tensorcnt(0);
    __syncthreads();
  }

  // C/D layout: VGPR r holds row m = r + 8*(lane>>4), column n = lane&15.
  const int m0 = blockIdx.x * 128 + wm * 64;
  const int n0 = blockIdx.y * 128 + wn * 64;
  const int cn = lane & 15;
  const int mh = (lane >> 4) << 3;
#pragma unroll
  for (int i = 0; i < 4; ++i) {
#pragma unroll
    for (int j = 0; j < 4; ++j) {
#pragma unroll
      for (int r = 0; r < 8; ++r) {
        const int m = m0 + i * 16 + mh + r;
        const int n = n0 + j * 16 + cn;
        float v = acc[i][j][r];
        if (HAS_BIAS) v += bias[n];
        if (OUT_BF16) {
          __bf16* C = (__bf16*)Cvoid + bz * strideC;
          if (TRANS_OUT) C[(size_t)n * ldc + m] = (__bf16)v;
          else           C[(size_t)m * ldc + n] = (__bf16)v;
        } else {
          float* C = (float*)Cvoid + bz * strideC;
          C[(size_t)m * ldc + n] = v;
        }
      }
    }
  }
}

// ---------------------------------------------------------------------------
// In-place row softmax over bf16 rows of 2048 (fp32 math, scale pre-applied).
// ---------------------------------------------------------------------------
__global__ __launch_bounds__(256) void softmax_rows(__bf16* __restrict__ P,
                                                    float scale) {
  const int tid = threadIdx.x;
  __bf16* row = P + (size_t)blockIdx.x * 2048;
  float v[8];
  float mx = -3.402823466e38f;
#pragma unroll
  for (int i = 0; i < 8; ++i) {
    v[i] = scale * (float)row[tid + (i << 8)];
    mx = fmaxf(mx, v[i]);
  }
#pragma unroll
  for (int off = 16; off > 0; off >>= 1)
    mx = fmaxf(mx, __shfl_xor(mx, off, 32));
  __shared__ float sred[8];
  const int w = tid >> 5;
  if ((tid & 31) == 0) sred[w] = mx;
  __syncthreads();
  float bm = sred[0];
#pragma unroll
  for (int i = 1; i < 8; ++i) bm = fmaxf(bm, sred[i]);
  __syncthreads();
  float sum = 0.0f;
#pragma unroll
  for (int i = 0; i < 8; ++i) {
    v[i] = __expf(v[i] - bm);
    sum += v[i];
  }
#pragma unroll
  for (int off = 16; off > 0; off >>= 1)
    sum += __shfl_xor(sum, off, 32);
  if ((tid & 31) == 0) sred[w] = sum;
  __syncthreads();
  float tot = 0.0f;
#pragma unroll
  for (int i = 0; i < 8; ++i) tot += sred[i];
  const float inv = 1.0f / tot;
#pragma unroll
  for (int i = 0; i < 8; ++i)
    row[tid + (i << 8)] = (__bf16)(v[i] * inv);
}

// ---------------------------------------------------------------------------
extern "C" void kernel_launch(void* const* d_in, const int* in_sizes, int n_in,
                              void* d_out, int out_size, void* d_ws,
                              size_t ws_size, hipStream_t stream) {
  (void)in_sizes; (void)n_in; (void)out_size; (void)ws_size;
  const float* x  = (const float*)d_in[0];
  const float* Wq = (const float*)d_in[1];
  const float* bq = (const float*)d_in[2];
  const float* Wk = (const float*)d_in[3];
  const float* bk = (const float*)d_in[4];
  const float* Wv = (const float*)d_in[5];
  const float* bv = (const float*)d_in[6];
  const float* Wo = (const float*)d_in[7];
  const float* bo = (const float*)d_in[8];

  constexpr int Bn = 4, S = 2048, D = 1024;
  constexpr int M = Bn * S;                  // 8192 flattened rows
  constexpr size_t XB = (size_t)M * D * 2;
  constexpr size_t WB = (size_t)D * D * 2;

  // Workspace (bf16): xb | Wqb Wkb Wvb Wob | Qb | Kb | Vt | P ; O reuses Qb.
  char* ws = (char*)d_ws;
  __bf16* xb  = (__bf16*)(ws);
  __bf16* Wqb = (__bf16*)(ws + XB);
  __bf16* Wkb = (__bf16*)(ws + XB + 1 * WB);
  __bf16* Wvb = (__bf16*)(ws + XB + 2 * WB);
  __bf16* Wob = (__bf16*)(ws + XB + 3 * WB);
  __bf16* Qb  = (__bf16*)(ws + XB + 4 * WB);
  __bf16* Kb  = (__bf16*)(ws + 2 * XB + 4 * WB);
  __bf16* Vt  = (__bf16*)(ws + 3 * XB + 4 * WB);
  __bf16* P   = (__bf16*)(ws + 4 * XB + 4 * WB);
  __bf16* Ob  = Qb;

  const dim3 blk(128);

  // one-shot fp32 -> bf16 conversions
  cvt_f32_bf16<<<dim3((M * D) / 2048), dim3(256), 0, stream>>>(x, xb);
  cvt_f32_bf16<<<dim3((D * D) / 2048), dim3(256), 0, stream>>>(Wq, Wqb);
  cvt_f32_bf16<<<dim3((D * D) / 2048), dim3(256), 0, stream>>>(Wk, Wkb);
  cvt_f32_bf16<<<dim3((D * D) / 2048), dim3(256), 0, stream>>>(Wv, Wvb);
  cvt_f32_bf16<<<dim3((D * D) / 2048), dim3(256), 0, stream>>>(Wo, Wob);

  // Q = x @ Wq^T + bq -> Qb [M, D]
  wmma_gemm_nt<true, false, true>
      <<<dim3(M / 128, D / 128, 1), blk, 0, stream>>>(
          xb, Wqb, bq, Qb, D, D, D, D, 0, 0, 0);
  // K = x @ Wk^T + bk -> Kb [M, D]
  wmma_gemm_nt<true, false, true>
      <<<dim3(M / 128, D / 128, 1), blk, 0, stream>>>(
          xb, Wkb, bk, Kb, D, D, D, D, 0, 0, 0);
  // V = x @ Wv^T + bv, stored transposed -> Vt [D, M] (ldc = M)
  wmma_gemm_nt<true, true, true>
      <<<dim3(M / 128, D / 128, 1), blk, 0, stream>>>(
          xb, Wvb, bv, Vt, D, D, D, M, 0, 0, 0);
  // scores = Q @ K^T (batched over B) -> P [B, S, S]
  wmma_gemm_nt<true, false, false>
      <<<dim3(S / 128, S / 128, Bn), blk, 0, stream>>>(
          Qb, Kb, nullptr, P, D, D, D, S,
          (long long)S * D, (long long)S * D, (long long)S * S);
  // softmax(scores / sqrt(D)) in place (scale = 1/32)
  softmax_rows<<<dim3(M), dim3(256), 0, stream>>>(P, 1.0f / 32.0f);
  // O = P @ V  (B operand = Vt rows, contiguous-K; batch offset = S columns)
  wmma_gemm_nt<true, false, false>
      <<<dim3(S / 128, D / 128, Bn), blk, 0, stream>>>(
          P, Vt, nullptr, Ob, S, S, M, D,
          (long long)S * S, (long long)S, (long long)S * D);
  // out = O @ Wo^T + bo -> fp32 d_out [M, D]
  wmma_gemm_nt<false, false, true>
      <<<dim3(M / 128, D / 128, 1), blk, 0, stream>>>(
          Ob, Wob, bo, (float*)d_out, D, D, D, D, 0, 0, 0);
}
